// AudioLiquidEmber_73134703116727
// MI455X (gfx1250) — compile-verified, matrix operand order checked by reading, not run on hardware
//
#include <hip/hip_runtime.h>
#include <hip/hip_bf16.h>

// ---------------- model constants ----------------
#define Bc   32
#define Tc   1024
#define Mc   128
#define Dc   512
#define Uc   512
#define DUc  1024   // D + U
#define H4c  2048   // 4*D
#define Lc   4
#define Cc   50
#define NWG  16     // scan cluster size (max workgroups per cluster)
#define SLC  32     // U/D columns owned per scan workgroup

typedef __attribute__((ext_vector_type(16))) __bf16 v16bf;
typedef __attribute__((ext_vector_type(8)))  float  v8f;

// ---------------- helpers ----------------
__device__ __forceinline__ float sigmoidf_(float x) { return 1.f / (1.f + __expf(-x)); }
__device__ __forceinline__ float geluf_(float x)    { return 0.5f * x * (1.f + erff(x * 0.70710678118654752f)); }

// Load a 16x32 bf16 WMMA fragment (A layout; B uses the same layout on a
// transposed [n][k] LDS tile). Per ISA 7.12.2:
//  lane<16 : row=lane,   K = 0..7  in v0..3, K = 16..23 in v4..7
//  lane>=16: row=lane-16,K = 8..15 in v0..3, K = 24..31 in v4..7
__device__ __forceinline__ v16bf load_frag16(const __bf16* base, int stride, int row0, int lane) {
    const __bf16* p = base + (size_t)(row0 + (lane & 15)) * stride + ((lane >> 4) ? 8 : 0);
    union { v16bf v; unsigned u[8]; } f;
#pragma unroll
    for (int i = 0; i < 4; ++i) f.u[i] = *(const unsigned*)(p + 2 * i);
#pragma unroll
    for (int i = 0; i < 4; ++i) f.u[4 + i] = *(const unsigned*)(p + 16 + 2 * i);
    return f.v;
}

// ---------------- fp32 -> bf16 convert (same layout) ----------------
__global__ void cvt_f32_bf16(const float* __restrict__ in, __bf16* __restrict__ out, int n) {
    for (int i = blockIdx.x * blockDim.x + threadIdx.x; i < n; i += gridDim.x * blockDim.x)
        out[i] = (__bf16)in[i];
}

// ---------------- fp32 [K,N] -> bf16 [N,K] tiled transpose-convert ----------
__global__ __launch_bounds__(256)
void cvt_transpose(const float* __restrict__ in, __bf16* __restrict__ out, int K, int N) {
    __shared__ float tile[32][33];
    const float* src = in + (size_t)blockIdx.z * K * N;
    __bf16* dst = out + (size_t)blockIdx.z * K * N;
    const int kb = blockIdx.y * 32, nb = blockIdx.x * 32;
    const int tx = threadIdx.x & 31, ty = threadIdx.x >> 5;   // 32 x 8
#pragma unroll
    for (int kk = ty; kk < 32; kk += 8) {
        int k = kb + kk, n = nb + tx;
        tile[kk][tx] = (k < K && n < N) ? src[(size_t)k * N + n] : 0.f;
    }
    __syncthreads();
#pragma unroll
    for (int nn = ty; nn < 32; nn += 8) {
        int n = nb + nn, k = kb + tx;
        if (n < N && k < K) dst[(size_t)n * K + k] = (__bf16)tile[tx][nn];
    }
}

// ---------------- generic WMMA GEMM: C = A[MrxK] @ B[KxN] (+ epilogue) ------
// B pre-transposed: Bt[N][K]. Block tile 128x128, 8 waves (2Mx4N), each wave
// 64x32 = 4x2 16x16 tiles, K stepped by 32. Double-buffered LDS staging:
// per k-tile: issue next global loads -> WMMA current -> store regs to other
// buffer -> one barrier (global latency hidden behind matrix ops).
// mode 0: Cf = acc + bias; mode 1: Cbf = gelu(acc+bias); mode 2: Cf = acc+bias+Cres
__global__ __launch_bounds__(256)
void wmma_gemm(const __bf16* __restrict__ A, const __bf16* __restrict__ Bt,
               const float* __restrict__ bias, const float* __restrict__ Cres,
               float* __restrict__ Cf, __bf16* __restrict__ Cbf,
               int Mr, int N, int K, int mode) {
    __shared__ __bf16 As[2][128 * 40];   // [m][k], padded stride 40
    __shared__ __bf16 Bs[2][128 * 40];   // [n][k], padded stride 40
    const int tid  = threadIdx.x, lane = tid & 31, wid = tid >> 5;
    const int wm = wid >> 2, wn = wid & 3;
    const int m0 = blockIdx.y * 128, n0 = blockIdx.x * 128;
    const int r = tid >> 1, half = (tid & 1) * 16;     // staging: half-row per thread
    const bool av = (m0 + r < Mr), bv = (n0 + r < N);
    const __bf16* arow = A  + (size_t)(m0 + r) * K + half;
    const __bf16* brow = Bt + (size_t)(n0 + r) * K + half;
    const int nkt = K / 32;

    uint4 ra0, ra1, rb0, rb1;
    uint4 z4; z4.x = z4.y = z4.z = z4.w = 0u;
    auto gload = [&](int k0) {
        if (av) { const uint4* s = (const uint4*)(arow + k0); ra0 = s[0]; ra1 = s[1]; }
        else    { ra0 = z4; ra1 = z4; }
        if (bv) { const uint4* s = (const uint4*)(brow + k0); rb0 = s[0]; rb1 = s[1]; }
        else    { rb0 = z4; rb1 = z4; }
    };
    auto sstore = [&](int buf) {
        uint4* da = (uint4*)&As[buf][r * 40 + half]; da[0] = ra0; da[1] = ra1;
        uint4* db = (uint4*)&Bs[buf][r * 40 + half]; db[0] = rb0; db[1] = rb1;
    };

    const v8f z8 = {0.f,0.f,0.f,0.f,0.f,0.f,0.f,0.f};
    v8f acc[4][2];
#pragma unroll
    for (int a = 0; a < 4; ++a)
#pragma unroll
        for (int b = 0; b < 2; ++b) acc[a][b] = z8;

    gload(0); sstore(0);
    __syncthreads();
    for (int kt = 0; kt < nkt; ++kt) {
        const int cur = kt & 1;
        if (kt + 1 < nkt) gload((kt + 1) * 32);
        if (kt + 2 < nkt) {                          // keep L2 warm -> global_prefetch_b8
            if (av) __builtin_prefetch(arow + (kt + 2) * 32, 0, 1);
            if (bv) __builtin_prefetch(brow + (kt + 2) * 32, 0, 1);
        }
#pragma unroll
        for (int mt = 0; mt < 4; ++mt) {
            v16bf af = load_frag16(As[cur], 40, wm * 64 + mt * 16, lane);
#pragma unroll
            for (int nt = 0; nt < 2; ++nt) {
                v16bf bf = load_frag16(Bs[cur], 40, wn * 32 + nt * 16, lane);
                acc[mt][nt] = __builtin_amdgcn_wmma_f32_16x16x32_bf16(
                    false, af, false, bf, (short)0, acc[mt][nt], false, false);
            }
        }
        if (kt + 1 < nkt) sstore(cur ^ 1);
        __syncthreads();
    }
    // epilogue (C/D layout: vgpr i -> row = i + (lane>=16?8:0), col = lane&15)
    const int colb = lane & 15, rowb = (lane >> 4) * 8;
#pragma unroll
    for (int mt = 0; mt < 4; ++mt)
#pragma unroll
        for (int nt = 0; nt < 2; ++nt)
#pragma unroll
            for (int i = 0; i < 8; ++i) {
                int row = m0 + wm * 64 + mt * 16 + rowb + i;
                int col = n0 + wn * 32 + nt * 16 + colb;
                if (row < Mr && col < N) {
                    float v = acc[mt][nt][i] + bias[col];
                    if (mode == 1)      Cbf[(size_t)row * N + col] = (__bf16)geluf_(v);
                    else if (mode == 2) Cf[(size_t)row * N + col]  = v + Cres[(size_t)row * N + col];
                    else                Cf[(size_t)row * N + col]  = v;
                }
            }
}

// ---------------- row LayerNorm (D=512), one wave32 per row, bf16 out -------
__global__ __launch_bounds__(256)
void ln_kernel(const float* __restrict__ x, const float* __restrict__ g,
               const float* __restrict__ b, __bf16* __restrict__ out, int nrows) {
    const int lane = threadIdx.x & 31, wid = threadIdx.x >> 5;
    const int row = blockIdx.x * 8 + wid;
    if (row >= nrows) return;
    const float* xr = x + (size_t)row * Dc;
    float vals[16], s = 0.f;
#pragma unroll
    for (int i = 0; i < 16; ++i) { vals[i] = xr[lane + 32 * i]; s += vals[i]; }
#pragma unroll
    for (int off = 16; off; off >>= 1) s += __shfl_xor(s, off, 32);
    const float mean = s * (1.f / Dc);
    float vs = 0.f;
#pragma unroll
    for (int i = 0; i < 16; ++i) { float d = vals[i] - mean; vs += d * d; }
#pragma unroll
    for (int off = 16; off; off >>= 1) vs += __shfl_xor(vs, off, 32);
    const float inv = rsqrtf(vs * (1.f / Dc) + 1e-5f);
#pragma unroll
    for (int i = 0; i < 16; ++i) {
        int d = lane + 32 * i;
        out[(size_t)row * Dc + d] = (__bf16)((vals[i] - mean) * inv * g[d] + b[d]);
    }
}

// ---------------- mean pool over T, bf16 out --------------------------------
__global__ void pool_kernel(const __bf16* __restrict__ xn, __bf16* __restrict__ pooled) {
    int i = blockIdx.x * blockDim.x + threadIdx.x;
    if (i >= Bc * Dc) return;
    int b = i / Dc, d = i % Dc;
    float s = 0.f;
    for (int t = 0; t < Tc; ++t) s += (float)xn[((size_t)b * Tc + t) * Dc + d];
    pooled[i] = (__bf16)(s * (1.f / Tc));
}

// ---------------- persistent CfC+LIF scan, one layer ------------------------
// 16-workgroup cluster; WG w owns U/D columns [32w,32w+32). Pre-transposed
// [n][k] weights; double-buffered LDS staging in both phases; one cluster
// barrier per step (ping-pong h), h exchanged through L2.
__global__ __launch_bounds__(256)
void scan_layer(const __bf16* __restrict__ xn,
                const __bf16* __restrict__ Wg0T, const __bf16* __restrict__ Wg1T,
                const __bf16* __restrict__ Wg2T, const __bf16* __restrict__ Wg3T,
                const float* __restrict__ bff1, const float* __restrict__ bff2,
                const float* __restrict__ bta,  const float* __restrict__ btb,
                const __bf16* __restrict__ WoutT, const float* __restrict__ bout,
                const float* __restrict__ thr, const float* __restrict__ leak,
                const float* __restrict__ steep,
                float* __restrict__ xf, __bf16* h0, __bf16* h1) {
    __shared__ __bf16 zs[2][Bc * 40];         // z / h tile: 32 rows x 32 k
    __shared__ __bf16 wt[2][4][SLC * 40];     // per-gate weight tile [n][k]
    __shared__ float  accs[4][Bc * SLC];      // accumulators 4 x 32 x 32

    const int tid = threadIdx.x, lane = tid & 31, wid = tid >> 5;
    const int wg = blockIdx.x;
    const int u0 = wg * SLC, d0 = wg * SLC;
    const __bf16* WgT[4] = {Wg0T, Wg1T, Wg2T, Wg3T};
    const v8f z8 = {0.f,0.f,0.f,0.f,0.f,0.f,0.f,0.f};

    // staging index sets (all vectorized):
    const int zb = tid >> 3, zk = (tid & 7) * 4;                 // 8B per thread
    const int wgg = tid >> 6, wu = (tid >> 1) & 31, wh = (tid & 1) * 16; // 32B per thread

    // per-thread LIF state: 4 elements e = tid*4+j of the 32x32 (b,d) slice
    float v[4], sleak[4], vthr[4], vsteep[4], vbout[4];
#pragma unroll
    for (int j = 0; j < 4; ++j) {
        int e = tid * 4 + j, dd = d0 + (e & 31);
        v[j] = 0.f; sleak[j] = sigmoidf_(leak[dd]);
        vthr[j] = thr[dd]; vsteep[j] = steep[dd]; vbout[j] = bout[dd];
    }
    for (int i = tid; i < Bc * SLC; i += 256)               // zero h0 slice
        h0[(i >> 5) * Uc + u0 + (i & 31)] = (__bf16)0.f;
    __syncthreads();
    __builtin_amdgcn_fence(__ATOMIC_RELEASE, "agent");
    __builtin_amdgcn_s_cluster_barrier();
    __builtin_amdgcn_fence(__ATOMIC_ACQUIRE, "agent");

    for (int t = 0; t < Tc; ++t) {
        const __bf16* hcur = (t & 1) ? h1 : h0;
        __bf16*       hnxt = (t & 1) ? h0 : h1;

        // -------- phase A: 4 gates, K = 1024, double-buffered --------
        const int g = wid >> 1, nt = wid & 1;   // wave job: (gate, n-tile)
        v8f acc[2]; acc[0] = z8; acc[1] = z8;
        {
            uint2 rz; uint4 rw0, rw1;
            auto loadA = [&](int k0) {
                const __bf16* zsrc = (k0 < Dc)                  // tile-uniform select
                    ? xn + ((size_t)zb * Tc + t) * Dc + k0 + zk
                    : hcur + (size_t)zb * Uc + (k0 - Dc) + zk;
                rz = *(const uint2*)zsrc;
                const uint4* s = (const uint4*)(WgT[wgg] + (size_t)(u0 + wu) * DUc + k0 + wh);
                rw0 = s[0]; rw1 = s[1];
            };
            auto storeA = [&](int buf) {
                *(uint2*)&zs[buf][zb * 40 + zk] = rz;
                uint4* d = (uint4*)&wt[buf][wgg][wu * 40 + wh];
                d[0] = rw0; d[1] = rw1;
            };
            loadA(0); storeA(0);
            __syncthreads();
            for (int kt = 0; kt < DUc / 32; ++kt) {
                const int cur = kt & 1;
                if (kt + 1 < DUc / 32) loadA((kt + 1) * 32);
                v16bf bf = load_frag16(&wt[cur][g][0], 40, nt * 16, lane);
#pragma unroll
                for (int mt = 0; mt < 2; ++mt) {
                    v16bf af = load_frag16(zs[cur], 40, mt * 16, lane);
                    acc[mt] = __builtin_amdgcn_wmma_f32_16x16x32_bf16(
                        false, af, false, bf, (short)0, acc[mt], false, false);
                }
                if (kt + 1 < DUc / 32) storeA(cur ^ 1);
                __syncthreads();
            }
        }
        {   // spill gate accs to LDS
            const int colb = lane & 15, rowb = (lane >> 4) * 8;
#pragma unroll
            for (int mt = 0; mt < 2; ++mt)
#pragma unroll
                for (int i = 0; i < 8; ++i)
                    accs[g][(mt * 16 + rowb + i) * SLC + nt * 16 + colb] = acc[mt][i];
        }
        __syncthreads();
        // gate nonlinearity -> h_new slice
#pragma unroll
        for (int j = 0; j < 4; ++j) {
            int e = tid * 4 + j, b = e >> 5, gu = u0 + (e & 31);
            float f1 = tanhf(accs[0][e] + bff1[gu]);
            float f2 = tanhf(accs[1][e] + bff2[gu]);
            float ti = sigmoidf_(accs[2][e] + bta[gu] + accs[3][e] + btb[gu]);
            hnxt[b * Uc + gu] = (__bf16)(f1 * (1.f - ti) + ti * f2);
        }
        __syncthreads();
        __builtin_amdgcn_fence(__ATOMIC_RELEASE, "agent");
        __builtin_amdgcn_s_cluster_barrier();
        __builtin_amdgcn_fence(__ATOMIC_ACQUIRE, "agent");

        // -------- phase B: o slice = h_new @ Wout[:,slice], K = 512 --------
        v8f oa = z8;
        const int bmt = wid & 1, bnt = (wid >> 1) & 1;
        {
            uint2 rh, rwo;
            auto loadB = [&](int k0) {
                rh  = *(const uint2*)(hnxt + (size_t)zb * Uc + k0 + zk);
                rwo = *(const uint2*)(WoutT + (size_t)(d0 + zb) * Uc + k0 + zk);
            };
            auto storeB = [&](int buf) {
                *(uint2*)&zs[buf][zb * 40 + zk]    = rh;
                *(uint2*)&wt[buf][0][zb * 40 + zk] = rwo;
            };
            loadB(0); storeB(0);
            __syncthreads();
            for (int kt = 0; kt < Uc / 32; ++kt) {
                const int cur = kt & 1;
                if (kt + 1 < Uc / 32) loadB((kt + 1) * 32);
                if (wid < 4) {
                    v16bf af = load_frag16(zs[cur], 40, bmt * 16, lane);
                    v16bf bf = load_frag16(&wt[cur][0][0], 40, bnt * 16, lane);
                    oa = __builtin_amdgcn_wmma_f32_16x16x32_bf16(
                        false, af, false, bf, (short)0, oa, false, false);
                }
                if (kt + 1 < Uc / 32) storeB(cur ^ 1);
                __syncthreads();
            }
        }
        if (wid < 4) {
            const int colb = lane & 15, rowb = (lane >> 4) * 8;
#pragma unroll
            for (int i = 0; i < 8; ++i)
                accs[0][(bmt * 16 + rowb + i) * SLC + bnt * 16 + colb] = oa[i];
        }
        __syncthreads();
        // LIF: leaky integrate, surrogate spike, soft reset; residual add
#pragma unroll
        for (int j = 0; j < 4; ++j) {
            int e = tid * 4 + j, b = e >> 5, gd = d0 + (e & 31);
            float o  = accs[0][e] + vbout[j];
            float vn = sleak[j] * v[j] + o;
            float s  = sigmoidf_(vsteep[j] * (vn - vthr[j]));
            v[j] = vn - s * vthr[j];
            xf[((size_t)b * Tc + t) * Dc + gd] += o * s;
        }
        __syncthreads();
    }
}

// ---------------- host side ----------------
extern "C" void kernel_launch(void* const* d_in, const int* in_sizes, int n_in,
                              void* d_out, int out_size, void* d_ws, size_t ws_size,
                              hipStream_t stream) {
    (void)in_sizes; (void)n_in; (void)out_size; (void)ws_size;
    const float* mel   = (const float*)d_in[0];
    const float* Wp    = (const float*)d_in[1];
    const float* bp    = (const float*)d_in[2];
    const float* ln1g  = (const float*)d_in[3];
    const float* ln1b  = (const float*)d_in[4];
    const float* Wff1  = (const float*)d_in[5];
    const float* bff1  = (const float*)d_in[6];
    const float* Wff2  = (const float*)d_in[7];
    const float* bff2  = (const float*)d_in[8];
    const float* Wta   = (const float*)d_in[9];
    const float* bta   = (const float*)d_in[10];
    const float* Wtb   = (const float*)d_in[11];
    const float* btb   = (const float*)d_in[12];
    const float* Wout  = (const float*)d_in[13];
    const float* bout  = (const float*)d_in[14];
    const float* thr   = (const float*)d_in[15];
    const float* leak  = (const float*)d_in[16];
    const float* steep = (const float*)d_in[17];
    const float* ln2g  = (const float*)d_in[18];
    const float* ln2b  = (const float*)d_in[19];
    const float* W1    = (const float*)d_in[20];
    const float* b1    = (const float*)d_in[21];
    const float* W2    = (const float*)d_in[22];
    const float* b2    = (const float*)d_in[23];
    const float* lnfg  = (const float*)d_in[24];
    const float* lnfb  = (const float*)d_in[25];
    const float* Wc    = (const float*)d_in[26];
    const float* bc    = (const float*)d_in[27];

    char* ws = (char*)d_ws;
    size_t cur = 0;
    auto alloc = [&](size_t bytes) { size_t r = cur; cur += (bytes + 255) & ~(size_t)255; return r; };

    __bf16* mel_bf  = (__bf16*)(ws + alloc((size_t)Bc*Tc*Mc*2));
    __bf16* WpT     = (__bf16*)(ws + alloc((size_t)Mc*Dc*2));        // [D, M]
    __bf16* Wff1T   = (__bf16*)(ws + alloc((size_t)Lc*DUc*Uc*2));    // [l][U, DU]
    __bf16* Wff2T   = (__bf16*)(ws + alloc((size_t)Lc*DUc*Uc*2));
    __bf16* WtaT    = (__bf16*)(ws + alloc((size_t)Lc*DUc*Uc*2));
    __bf16* WtbT    = (__bf16*)(ws + alloc((size_t)Lc*DUc*Uc*2));
    __bf16* WoutT   = (__bf16*)(ws + alloc((size_t)Lc*Uc*Dc*2));     // [l][D, U]
    __bf16* W1T     = (__bf16*)(ws + alloc((size_t)Lc*Dc*H4c*2));    // [l][4D, D]
    __bf16* W2T     = (__bf16*)(ws + alloc((size_t)Lc*H4c*Dc*2));    // [l][D, 4D]
    __bf16* WcT     = (__bf16*)(ws + alloc((size_t)Dc*Cc*2));        // [C, D]
    float*  xf      = (float*)(ws + alloc((size_t)Bc*Tc*Dc*4));
    __bf16* xn_bf   = (__bf16*)(ws + alloc((size_t)Bc*Tc*Dc*2));
    __bf16* act_bf  = (__bf16*)(ws + alloc((size_t)Bc*Tc*H4c*2));
    __bf16* h0      = (__bf16*)(ws + alloc((size_t)Bc*Uc*2));
    __bf16* h1      = (__bf16*)(ws + alloc((size_t)Bc*Uc*2));
    __bf16* pool_bf = (__bf16*)(ws + alloc((size_t)Bc*Dc*2));

    // one-time convert / transpose-convert of all operands
    cvt_f32_bf16<<<1024, 256, 0, stream>>>(mel, mel_bf, Bc*Tc*Mc);
    cvt_transpose<<<dim3(Dc/32,  Mc/32,  1),  256, 0, stream>>>(Wp,   WpT,   Mc,  Dc);
    cvt_transpose<<<dim3(Uc/32,  DUc/32, Lc), 256, 0, stream>>>(Wff1, Wff1T, DUc, Uc);
    cvt_transpose<<<dim3(Uc/32,  DUc/32, Lc), 256, 0, stream>>>(Wff2, Wff2T, DUc, Uc);
    cvt_transpose<<<dim3(Uc/32,  DUc/32, Lc), 256, 0, stream>>>(Wta,  WtaT,  DUc, Uc);
    cvt_transpose<<<dim3(Uc/32,  DUc/32, Lc), 256, 0, stream>>>(Wtb,  WtbT,  DUc, Uc);
    cvt_transpose<<<dim3(Dc/32,  Uc/32,  Lc), 256, 0, stream>>>(Wout, WoutT, Uc,  Dc);
    cvt_transpose<<<dim3(H4c/32, Dc/32,  Lc), 256, 0, stream>>>(W1,   W1T,   Dc,  H4c);
    cvt_transpose<<<dim3(Dc/32,  H4c/32, Lc), 256, 0, stream>>>(W2,   W2T,   H4c, Dc);
    cvt_transpose<<<dim3((Cc+31)/32, Dc/32, 1), 256, 0, stream>>>(Wc, WcT,   Dc,  Cc);

    const int ROWS = Bc * Tc;   // 32768

    // x = mel @ Wp + bp
    wmma_gemm<<<dim3(Dc/128, ROWS/128), 256, 0, stream>>>(
        mel_bf, WpT, bp, nullptr, xf, nullptr, ROWS, Dc, Mc, 0);

    for (int l = 0; l < Lc; ++l) {
        ln_kernel<<<ROWS/8, 256, 0, stream>>>(xf, ln1g + l*Dc, ln1b + l*Dc, xn_bf, ROWS);
        scan_layer<<<NWG, 256, 0, stream>>>(
            xn_bf,
            Wff1T + (size_t)l*DUc*Uc, Wff2T + (size_t)l*DUc*Uc,
            WtaT  + (size_t)l*DUc*Uc, WtbT  + (size_t)l*DUc*Uc,
            bff1 + l*Uc, bff2 + l*Uc, bta + l*Uc, btb + l*Uc,
            WoutT + (size_t)l*Uc*Dc, bout + l*Dc,
            thr + l*Dc, leak + l*Dc, steep + l*Dc,
            xf, h0, h1);
        ln_kernel<<<ROWS/8, 256, 0, stream>>>(xf, ln2g + l*Dc, ln2b + l*Dc, xn_bf, ROWS);
        wmma_gemm<<<dim3(H4c/128, ROWS/128), 256, 0, stream>>>(
            xn_bf, W1T + (size_t)l*Dc*H4c, b1 + l*H4c, nullptr,
            nullptr, act_bf, ROWS, H4c, Dc, 1);
        wmma_gemm<<<dim3(Dc/128, ROWS/128), 256, 0, stream>>>(
            act_bf, W2T + (size_t)l*H4c*Dc, b2 + l*Dc, xf,
            xf, nullptr, ROWS, Dc, H4c, 2);
    }

    ln_kernel<<<ROWS/8, 256, 0, stream>>>(xf, lnfg, lnfb, xn_bf, ROWS);
    pool_kernel<<<(Bc*Dc + 255)/256, 256, 0, stream>>>(xn_bf, pool_bf);
    wmma_gemm<<<dim3(1, 1), 256, 0, stream>>>(
        pool_bf, WcT, bc, nullptr, (float*)d_out, nullptr, Bc, Cc, Dc, 0);
}
